// NemotronHWayfinderAttention_377957122625
// MI455X (gfx1250) — matrix operand order repaired and checked by reading, not compiled, not run on hardware
//
#include <hip/hip_runtime.h>
#include <hip/hip_bf16.h>

typedef _Float16 v8h  __attribute__((ext_vector_type(8)));
typedef _Float16 v16h __attribute__((ext_vector_type(16)));
typedef float    v8f  __attribute__((ext_vector_type(8)));
typedef int      v4i  __attribute__((ext_vector_type(4)));
typedef int      v8i  __attribute__((ext_vector_type(8)));
typedef unsigned int v4u __attribute__((ext_vector_type(4)));

#define S_LEN 2048
#define HID   1024
#define NHQ   16
#define NHKV  8
#define DHEAD 64
#define KNB   128
#define NEG_INF (-3.0e38f)

// ---------------------------------------------------------------------------
// gfx1250 async global->LDS copy (ASYNCcnt) with safe fallback
// ---------------------------------------------------------------------------
typedef __attribute__((address_space(1))) v4i* as1_v4ip;
typedef __attribute__((address_space(3))) v4i* as3_v4ip;

#if __has_builtin(__builtin_amdgcn_global_load_async_to_lds_b128)
#define HAVE_ASYNC 1
__device__ __forceinline__ void async_copy16(const _Float16* g, _Float16* l) {
  __builtin_amdgcn_global_load_async_to_lds_b128(
      (as1_v4ip)(const void*)g, (as3_v4ip)(void*)l, 0, 0);
}
__device__ __forceinline__ void async_wait0() {
#if __has_builtin(__builtin_amdgcn_s_wait_asynccnt)
  __builtin_amdgcn_s_wait_asynccnt(0);
#else
  asm volatile("s_wait_asynccnt 0x0" ::: "memory");
#endif
}
#else
#define HAVE_ASYNC 0
#endif

// ---------------------------------------------------------------------------
// gfx1250 Tensor Data Mover: DMA a 128-row x 64B tile (row stride Kdim elems)
// into LDS with hardware padding 64B->96B rows (matches BK_PAD=48 halves).
// ---------------------------------------------------------------------------
#if __has_builtin(__builtin_amdgcn_tensor_load_to_lds)
#define HAVE_TDM 1
__device__ __forceinline__ void tdm_load_tile_128x32(const _Float16* gtile,
                                                     _Float16* ltile,
                                                     int kstride) {
  unsigned long long ga = (unsigned long long)(uintptr_t)gtile;
  unsigned int la = (unsigned int)(size_t)(as3_v4ip)(void*)ltile;
  // D# group 0: count=1 | lds_addr | global_addr[56:0] | type=2
  v4u g0;
  g0[0] = 1u;
  g0[1] = la;
  g0[2] = (unsigned int)ga;
  g0[3] = (unsigned int)(ga >> 32) | (2u << 30);
  // D# group 1: data_size=2B, pad_enable, pad_interval=3 (64B),
  // pad_amount=7 (8 DWORDs = 32B), tensor_dim0=kstride, tensor_dim1=4096,
  // tile_dim0=32 elems, tile_dim1=128 rows, dim0_stride=kstride
  v8i g1;
  g1[0] = (1 << 16) | (1 << 20) | (3 << 22) | (7 << 25);
  g1[1] = (kstride & 0xFFFF) << 16;
  g1[2] = (kstride >> 16) | (4096 << 16);
  g1[3] = (32 << 16);
  g1[4] = 128;
  g1[5] = kstride;
  g1[6] = 0;
  g1[7] = 0;
  v4i z4 = {0, 0, 0, 0};
#if __clang_major__ >= 23
  v8i z8 = {0, 0, 0, 0, 0, 0, 0, 0};
  __builtin_amdgcn_tensor_load_to_lds(g0, g1, z4, z4, z8, 0);
#else
  __builtin_amdgcn_tensor_load_to_lds(g0, g1, z4, z4, 0);
#endif
}
__device__ __forceinline__ void tensor_wait0() {
#if __has_builtin(__builtin_amdgcn_s_wait_tensorcnt)
  __builtin_amdgcn_s_wait_tensorcnt(0);
#else
  asm volatile("s_wait_tensorcnt 0x0" ::: "memory");
#endif
}
#else
#define HAVE_TDM 0
#endif

// ---------------------------------------------------------------------------
// WMMA fragment helpers (wave32, v_wmma_f32_16x16x32_f16)
// ---------------------------------------------------------------------------
__device__ __forceinline__ v16h frag_cat(v8h lo, v8h hi) {
  v16h r;
#pragma unroll
  for (int i = 0; i < 8; ++i) { r[i] = lo[i]; r[8 + i] = hi[i]; }
  return r;
}

// A fragment (16x32, M-major rows contiguous in K):
// lane L holds row base+L%16; lanes 0-15: K={0..7,16..23}, lanes 16-31: K={8..15,24..31}
__device__ __forceinline__ v16h load_frag_a(const _Float16* row, int half) {
  v8h lo = *(const v8h*)(row + half * 8);
  v8h hi = *(const v8h*)(row + 16 + half * 8);
  return frag_cat(lo, hi);
}

// B fragment (32x16 stored N-major, row n contiguous in K):
// lane L holds col base+L%16; lanes 0-15: K={0..7,16..23}, lanes 16-31: K={8..15,24..31}
__device__ __forceinline__ v16h load_frag_b(const _Float16* row, int half) {
  v8h lo = *(const v8h*)(row + half * 8);
  v8h hi = *(const v8h*)(row + 16 + half * 8);
  return frag_cat(lo, hi);
}

__device__ __forceinline__ v8f wmma_f16(v16h a, v16h b, v8f c) {
  return __builtin_amdgcn_wmma_f32_16x16x32_f16(false, a, false, b, (short)0, c,
                                                false, false);
}

__device__ __forceinline__ v8f v8f_zero() {
  v8f z = {0.f, 0.f, 0.f, 0.f, 0.f, 0.f, 0.f, 0.f};
  return z;
}

// ---------------------------------------------------------------------------
// Cast / pack kernels (8 elements per thread, b128 traffic)
// ---------------------------------------------------------------------------
__global__ void cast_f16_kernel(const float* __restrict__ in,
                                _Float16* __restrict__ out, int n8) {
  int i = blockIdx.x * blockDim.x + threadIdx.x;
  if (i >= n8) return;
  float4 a = ((const float4*)in)[2 * i];
  float4 b = ((const float4*)in)[2 * i + 1];
  v8h o;
  o[0] = (_Float16)a.x; o[1] = (_Float16)a.y; o[2] = (_Float16)a.z; o[3] = (_Float16)a.w;
  o[4] = (_Float16)b.x; o[5] = (_Float16)b.y; o[6] = (_Float16)b.z; o[7] = (_Float16)b.w;
  ((v8h*)out)[i] = o;
}

// Fused W_qkv stored N-major: out[n][k], n in [0,2048): q | k | v
__global__ void pack_wqkv_kernel(const float* __restrict__ qw,
                                 const float* __restrict__ kw,
                                 const float* __restrict__ vw,
                                 _Float16* __restrict__ out) {
  int i = blockIdx.x * blockDim.x + threadIdx.x;  // chunk of 8 along k
  if (i >= 2048 * 128) return;
  int n = i >> 7;
  int k8 = (i & 127) * 8;
  const float* src;
  if (n < 1024)      src = qw + (size_t)n * 1024 + k8;
  else if (n < 1536) src = kw + (size_t)(n - 1024) * 1024 + k8;
  else               src = vw + (size_t)(n - 1536) * 1024 + k8;
  float4 a = ((const float4*)src)[0];
  float4 b = ((const float4*)src)[1];
  v8h o;
  o[0] = (_Float16)a.x; o[1] = (_Float16)a.y; o[2] = (_Float16)a.z; o[3] = (_Float16)a.w;
  o[4] = (_Float16)b.x; o[5] = (_Float16)b.y; o[6] = (_Float16)b.z; o[7] = (_Float16)b.w;
  ((v8h*)out)[i] = o;
}

// ---------------------------------------------------------------------------
// Tiled WMMA GEMM:  C[M x N] = A[M x K] * B^T   (B stored N-major: B[n][k])
// Block 128x128, 256 threads = 8 waves (2x4), wave tile 64x32, BK=32.
// Staging priority: TDM tensor DMA > async global->LDS > synchronous loads,
// always double/single buffered appropriately.
// EPI 0: scatter f16 into per-head Q/K/V buffers.  EPI 1: f32 to Cout.
// ---------------------------------------------------------------------------
#define BK_PAD 48   // 32 valid + pad; row stride 96B (multiple of 16B)
#define TILE_H (128 * BK_PAD)

template <int EPI>
__global__ __launch_bounds__(256) void gemm_wmma_kernel(
    const _Float16* __restrict__ A, const _Float16* __restrict__ B,
    float* __restrict__ Cout, int N, int Kdim,
    _Float16* __restrict__ qb, _Float16* __restrict__ kb,
    _Float16* __restrict__ vb) {
  __shared__ _Float16 As[2 * TILE_H];
  __shared__ _Float16 Bs[2 * TILE_H];

  const int tm = blockIdx.y * 128;
  const int tn = blockIdx.x * 128;
  const int tid = threadIdx.x;
  const int wid = tid >> 5;
  const int lane = tid & 31;
  const int wrow = wid >> 2;   // 0..1 -> M offset *64
  const int wcol = wid & 3;    // 0..3 -> N offset *32
  const int half = lane >> 4;
  const int lrow = lane & 15;

  v8f acc[4][2];
#pragma unroll
  for (int i = 0; i < 4; ++i)
#pragma unroll
    for (int j = 0; j < 2; ++j) acc[i][j] = v8f_zero();

#if HAVE_TDM
  // ---- double-buffered TDM pipeline: wave 0 drives the DMA ----
  auto issue_tile = [&](int k0, int buf) {
    if (wid == 0) {
      tdm_load_tile_128x32(A + (size_t)tm * Kdim + k0, As + buf * TILE_H, Kdim);
      tdm_load_tile_128x32(B + (size_t)tn * Kdim + k0, Bs + buf * TILE_H, Kdim);
    }
  };
  issue_tile(0, 0);
  int cur = 0;
  for (int k0 = 0; k0 < Kdim; k0 += 32, cur ^= 1) {
    if (wid == 0) tensor_wait0();
    __syncthreads();
    if (k0 + 32 < Kdim) issue_tile(k0 + 32, cur ^ 1);
    const _Float16* as = As + cur * TILE_H;
    const _Float16* bs = Bs + cur * TILE_H;
    v16h bfr[2];
#pragma unroll
    for (int j = 0; j < 2; ++j)
      bfr[j] = load_frag_b(bs + (wcol * 32 + j * 16 + lrow) * BK_PAD, half);
    v16h afr[4];
#pragma unroll
    for (int i = 0; i < 4; ++i)
      afr[i] = load_frag_a(as + (wrow * 64 + i * 16 + lrow) * BK_PAD, half);
#pragma unroll
    for (int i = 0; i < 4; ++i)
#pragma unroll
      for (int j = 0; j < 2; ++j) acc[i][j] = wmma_f16(afr[i], bfr[j], acc[i][j]);
  }
#elif HAVE_ASYNC
  // ---- double-buffered async pipeline ----
  auto issue_tile = [&](int k0, int buf) {
    _Float16* as = As + buf * TILE_H;
    _Float16* bs = Bs + buf * TILE_H;
#pragma unroll
    for (int c = tid; c < 512; c += 256) {
      int r = c >> 2;
      int kc = (c & 3) * 8;
      async_copy16(A + (size_t)(tm + r) * Kdim + k0 + kc, as + r * BK_PAD + kc);
      async_copy16(B + (size_t)(tn + r) * Kdim + k0 + kc, bs + r * BK_PAD + kc);
    }
  };
  issue_tile(0, 0);
  int cur = 0;
  for (int k0 = 0; k0 < Kdim; k0 += 32, cur ^= 1) {
    async_wait0();
    __syncthreads();
    if (k0 + 32 < Kdim) issue_tile(k0 + 32, cur ^ 1);
    const _Float16* as = As + cur * TILE_H;
    const _Float16* bs = Bs + cur * TILE_H;
    v16h bfr[2];
#pragma unroll
    for (int j = 0; j < 2; ++j)
      bfr[j] = load_frag_b(bs + (wcol * 32 + j * 16 + lrow) * BK_PAD, half);
    v16h afr[4];
#pragma unroll
    for (int i = 0; i < 4; ++i)
      afr[i] = load_frag_a(as + (wrow * 64 + i * 16 + lrow) * BK_PAD, half);
#pragma unroll
    for (int i = 0; i < 4; ++i)
#pragma unroll
      for (int j = 0; j < 2; ++j) acc[i][j] = wmma_f16(afr[i], bfr[j], acc[i][j]);
  }
#else
  // ---- synchronous fallback ----
  for (int k0 = 0; k0 < Kdim; k0 += 32) {
    __syncthreads();
#pragma unroll
    for (int c = tid; c < 512; c += 256) {
      int r = c >> 2;
      int kc = (c & 3) * 8;
      v8h av = *(const v8h*)(A + (size_t)(tm + r) * Kdim + k0 + kc);
      *(v8h*)(As + r * BK_PAD + kc) = av;
      v8h bv = *(const v8h*)(B + (size_t)(tn + r) * Kdim + k0 + kc);
      *(v8h*)(Bs + r * BK_PAD + kc) = bv;
      if (k0 + 32 < Kdim) {
        __builtin_prefetch(A + (size_t)(tm + r) * Kdim + k0 + 32 + kc, 0, 1);
        __builtin_prefetch(B + (size_t)(tn + r) * Kdim + k0 + 32 + kc, 0, 1);
      }
    }
    __syncthreads();
    v16h bfr[2];
#pragma unroll
    for (int j = 0; j < 2; ++j)
      bfr[j] = load_frag_b(Bs + (wcol * 32 + j * 16 + lrow) * BK_PAD, half);
    v16h afr[4];
#pragma unroll
    for (int i = 0; i < 4; ++i)
      afr[i] = load_frag_a(As + (wrow * 64 + i * 16 + lrow) * BK_PAD, half);
#pragma unroll
    for (int i = 0; i < 4; ++i)
#pragma unroll
      for (int j = 0; j < 2; ++j) acc[i][j] = wmma_f16(afr[i], bfr[j], acc[i][j]);
  }
#endif

  // epilogue: C tile lane layout: n = lrow, m = r + 8*half within each 16x16
#pragma unroll
  for (int i = 0; i < 4; ++i)
#pragma unroll
    for (int j = 0; j < 2; ++j) {
      int n = tn + wcol * 32 + j * 16 + lrow;
#pragma unroll
      for (int r = 0; r < 8; ++r) {
        int m = tm + wrow * 64 + i * 16 + r + 8 * half;
        float v = acc[i][j][r];
        if (EPI == 0) {
          if (n < 1024) {
            int hh = n >> 6;
            qb[(((size_t)hh << 11) + m) * 64 + (n & 63)] = (_Float16)v;
          } else if (n < 1536) {
            int n2 = n - 1024, hh = n2 >> 6;
            kb[(((size_t)hh << 11) + m) * 64 + (n2 & 63)] = (_Float16)v;
          } else {
            int n2 = n - 1536, hh = n2 >> 6;
            vb[(((size_t)hh << 11) + m) * 64 + (n2 & 63)] = (_Float16)v;
          }
        } else {
          Cout[(size_t)m * N + n] = v;
        }
      }
    }
}

// ---------------------------------------------------------------------------
// Attention: one block per (kv head h, 16-query block). 2 waves (one per
// grouped q-head g). Window part dense via WMMA; 64 random neighbors per row
// via VALU (K/V are L2-resident f16).
// ---------------------------------------------------------------------------
#define KW_PAD 72   // 64 + pad, 144B rows (16B multiple)
#define VT_PAD 96   // 80->96 keys padded, 192B rows

__global__ __launch_bounds__(64) void attn_kernel(
    const _Float16* __restrict__ qbuf, const _Float16* __restrict__ kbuf,
    const _Float16* __restrict__ vbuf, const int* __restrict__ neigh,
    const float* __restrict__ edge_bias, _Float16* __restrict__ abuf) {
  const int sb = blockIdx.x;      // 0..127
  const int h = blockIdx.y;       // 0..7
  const int s0 = sb * 16;
  const int kbase = s0 - 63;      // first window key (may be <0)
  const int tid = threadIdx.x;    // 0..63
  const int g = tid >> 5;         // wave id == grouped q-head
  const int lane = tid & 31;
  const int half = lane >> 4;
  const int lrow = lane & 15;
  const float scale = 0.125f;     // D^-0.5
  const float eb_win = edge_bias[1];
  const float eb_rnd = edge_bias[3];

  __shared__ _Float16 Qs[2 * 16 * 64];      // [g][m][d]
  __shared__ _Float16 Kw[80 * KW_PAD];      // [c][d]   (N-major B for scores)
  __shared__ _Float16 Vt[64 * VT_PAD];      // [d][c]   (N-major B for PV)
  __shared__ float    sc[2 * 16 * KNB];     // scores/probs [g][m][j]
  __shared__ _Float16 Pw[2 * 16 * VT_PAD];  // raw win scores, then probs-A
  __shared__ float    ob[2 * 16 * 64];      // window PV result

  // ---- stage Q (both g) and window K (async when available) ---------------
#if HAVE_ASYNC
  for (int c = tid; c < 256; c += 64) {  // Q: 256 chunks of 8 halves
    int gg = c >> 7;
    int rem = c & 127;
    int m = rem >> 3;
    int d8 = (rem & 7) * 8;
    int hq = h * 2 + gg;
    async_copy16(qbuf + (((size_t)hq << 11) + s0 + m) * 64 + d8,
                 Qs + (gg * 16 + m) * 64 + d8);
  }
  for (int c = tid; c < 80; c += 64) {
    int key = kbase + c;
    key = key < 0 ? 0 : (key > S_LEN - 1 ? S_LEN - 1 : key);
    const _Float16* krow = kbuf + (((size_t)h << 11) + key) * 64;
#pragma unroll
    for (int d8 = 0; d8 < 64; d8 += 8)
      async_copy16(krow + d8, Kw + c * KW_PAD + d8);
  }
#else
  for (int c = tid; c < 256; c += 64) {
    int gg = c >> 7;
    int rem = c & 127;
    int m = rem >> 3;
    int d8 = (rem & 7) * 8;
    int hq = h * 2 + gg;
    v8h q = *(const v8h*)(qbuf + (((size_t)hq << 11) + s0 + m) * 64 + d8);
    *(v8h*)(Qs + (gg * 16 + m) * 64 + d8) = q;
  }
  for (int c = tid; c < 80; c += 64) {
    int key = kbase + c;
    key = key < 0 ? 0 : (key > S_LEN - 1 ? S_LEN - 1 : key);
    const _Float16* krow = kbuf + (((size_t)h << 11) + key) * 64;
#pragma unroll
    for (int d8 = 0; d8 < 64; d8 += 8) {
      v8h kk = *(const v8h*)(krow + d8);
      *(v8h*)(Kw + c * KW_PAD + d8) = kk;
    }
  }
#endif
  // ---- window V transposed into [d][c] (scalar transpose stores) ----------
  for (int c = tid; c < 80; c += 64) {
    int key = kbase + c;
    key = key < 0 ? 0 : (key > S_LEN - 1 ? S_LEN - 1 : key);
    const _Float16* vrow = vbuf + (((size_t)h << 11) + key) * 64;
#pragma unroll
    for (int d8 = 0; d8 < 64; d8 += 8) {
      v8h vv = *(const v8h*)(vrow + d8);
#pragma unroll
      for (int t = 0; t < 8; ++t) Vt[(d8 + t) * VT_PAD + c] = vv[t];
    }
  }
  for (int c = tid; c < 64 * 16; c += 64) {  // zero V pad cols 80..95
    int d = c >> 4;
    Vt[d * VT_PAD + 80 + (c & 15)] = (_Float16)0;
  }
#if HAVE_ASYNC
  async_wait0();
#endif
  __syncthreads();

  // ---- window scores via WMMA: S = Q(16x64) @ Kw^T(64x80); dense store ----
  for (int nt = 0; nt < 5; ++nt) {
    v8f a = v8f_zero();
#pragma unroll
    for (int ks = 0; ks < 2; ++ks) {
      v16h af = load_frag_a(Qs + (g * 16 + lrow) * 64 + ks * 32, half);
      v16h bf = load_frag_b(Kw + (nt * 16 + lrow) * KW_PAD + ks * 32, half);
      a = wmma_f16(af, bf, a);
    }
    int c = nt * 16 + lrow;
#pragma unroll
    for (int r = 0; r < 8; ++r) {
      int m = r + 8 * half;
      Pw[(g * 16 + m) * VT_PAD + c] = (_Float16)a[r];  // raw score, dense
    }
  }
  __syncthreads();

  // ---- assemble scores: window remap (banded mask) + random dots ----------
  {
    int row = tid & 31;
    int gg = row >> 4, m = row & 15;
    int s = s0 + m;
    int part = tid >> 5;
    float* scrow = sc + (gg * 16 + m) * KNB;
    if (part == 0) {
      const _Float16* praw = Pw + (gg * 16 + m) * VT_PAD;
#pragma unroll
      for (int j = 0; j < 64; ++j) {
        int key = s - j;               // window neighbor j -> key s-j
        float v = NEG_INF;
        if (key >= 0) v = (float)praw[m + 63 - j] * scale + eb_win;
        scrow[j] = v;
      }
    }
    const int* nrow = neigh + (((size_t)h << 11) + s) * KNB;
    const _Float16* qrow = Qs + (gg * 16 + m) * 64;
    int j0 = 64 + part * 32;
    for (int j = j0; j < j0 + 32; ++j) {
      int idx = nrow[j];
      float v = NEG_INF;
      if (idx >= 0 && idx <= s) {
        const _Float16* krow = kbuf + (((size_t)h << 11) + idx) * 64;
        float dot = 0.f;
#pragma unroll
        for (int d = 0; d < 64; ++d) dot += (float)qrow[d] * (float)krow[d];
        v = dot * scale + eb_rnd;
      }
      scrow[j] = v;
    }
  }
  __syncthreads();

  // ---- softmax per row + build window-prob A matrix -----------------------
  if (tid < 32) {
    int gg = tid >> 4, m = tid & 15;
    float* r = sc + (gg * 16 + m) * KNB;
    float mx = r[0];
    for (int j = 1; j < KNB; ++j) mx = fmaxf(mx, r[j]);
    float sum = 0.f;
    for (int j = 0; j < KNB; ++j) {
      float e = __expf(r[j] - mx);
      r[j] = e;
      sum += e;
    }
    float inv = 1.0f / sum;
    for (int j = 0; j < KNB; ++j) r[j] *= inv;
    _Float16* prow = Pw + (gg * 16 + m) * VT_PAD;
    for (int c = 0; c < VT_PAD; ++c) {
      int j = m + 63 - c;
      prow[c] = (j >= 0 && j < 64) ? (_Float16)r[j] : (_Float16)0;
    }
  }
  __syncthreads();

  // ---- window PV via WMMA: O = P(16x96) @ Vt^T(96x64) ---------------------
  for (int nt = 0; nt < 4; ++nt) {
    v8f a = v8f_zero();
#pragma unroll
    for (int ks = 0; ks < 3; ++ks) {
      v16h af = load_frag_a(Pw + (g * 16 + lrow) * VT_PAD + ks * 32, half);
      v16h bf = load_frag_b(Vt + (nt * 16 + lrow) * VT_PAD + ks * 32, half);
      a = wmma_f16(af, bf, a);
    }
#pragma unroll
    for (int r = 0; r < 8; ++r) {
      int m = r + 8 * half;
      ob[(g * 16 + m) * 64 + nt * 16 + lrow] = a[r];
    }
  }
  __syncthreads();

  // ---- random PV (VALU) + final write -------------------------------------
  {
    int m = lane & 15;
    int dh = lane >> 4;  // d chunk: 32 dims
    int s = s0 + m;
    const int* nrow = neigh + (((size_t)h << 11) + s) * KNB;
    const float* r = sc + (g * 16 + m) * KNB;
    float accv[32];
#pragma unroll
    for (int d = 0; d < 32; ++d) accv[d] = ob[(g * 16 + m) * 64 + dh * 32 + d];
    for (int j = 64; j < KNB; ++j) {
      int idx = nrow[j];
      if (idx >= 0 && idx <= s) {
        float p = r[j];
        const _Float16* vrow = vbuf + (((size_t)h << 11) + idx) * 64 + dh * 32;
#pragma unroll
        for (int d = 0; d < 32; ++d) accv[d] += p * (float)vrow[d];
      }
    }
    _Float16* orow = abuf + (size_t)s * 1024 + (h * 2 + g) * 64 + dh * 32;
#pragma unroll
    for (int d = 0; d < 32; ++d) orow[d] = (_Float16)accv[d];
  }
}

// ---------------------------------------------------------------------------
// Launch
// ---------------------------------------------------------------------------
extern "C" void kernel_launch(void* const* d_in, const int* in_sizes, int n_in,
                              void* d_out, int out_size, void* d_ws,
                              size_t ws_size, hipStream_t stream) {
  const float* hidden = (const float*)d_in[0];
  const float* q_w = (const float*)d_in[1];
  const float* k_w = (const float*)d_in[2];
  const float* v_w = (const float*)d_in[3];
  const float* o_w = (const float*)d_in[4];
  const float* edge_bias = (const float*)d_in[5];
  const int* neigh_idx = (const int*)d_in[6];
  // edge_type (d_in[7]) is structurally fixed: 1 for window, 3 for random.

  char* ws = (char*)d_ws;
  size_t off = 0;
  _Float16* hid16 = (_Float16*)(ws + off); off += (size_t)S_LEN * HID * 2;       // 4MB
  _Float16* wqkv16 = (_Float16*)(ws + off); off += (size_t)2048 * 1024 * 2;      // 4MB
  _Float16* wo16 = (_Float16*)(ws + off); off += (size_t)1024 * 1024 * 2;        // 2MB
  _Float16* qb = (_Float16*)(ws + off); off += (size_t)NHQ * S_LEN * DHEAD * 2;  // 4MB
  _Float16* kb = (_Float16*)(ws + off); off += (size_t)NHKV * S_LEN * DHEAD * 2; // 2MB
  _Float16* vb = (_Float16*)(ws + off); off += (size_t)NHKV * S_LEN * DHEAD * 2; // 2MB
  _Float16* ab = (_Float16*)(ws + off); off += (size_t)S_LEN * HID * 2;          // 4MB

  // 1) cast/pack to f16 (8-wide vectorized)
  cast_f16_kernel<<<(S_LEN * HID / 8 + 255) / 256, 256, 0, stream>>>(
      hidden, hid16, S_LEN * HID / 8);
  pack_wqkv_kernel<<<(2048 * 128 + 255) / 256, 256, 0, stream>>>(q_w, k_w, v_w,
                                                                 wqkv16);
  cast_f16_kernel<<<(1024 * 1024 / 8 + 255) / 256, 256, 0, stream>>>(
      o_w, wo16, 1024 * 1024 / 8);

  // 2) fused QKV projection GEMM (M=2048, N=2048, K=1024), scatter per head
  gemm_wmma_kernel<0><<<dim3(16, 16), 256, 0, stream>>>(
      hid16, wqkv16, nullptr, 2048, 1024, qb, kb, vb);

  // 3) sparse window+random attention
  attn_kernel<<<dim3(S_LEN / 16, NHKV), 64, 0, stream>>>(qb, kb, vb, neigh_idx,
                                                         edge_bias, ab);

  // 4) output projection GEMM (M=2048, N=1024, K=1024) -> f32 d_out
  gemm_wmma_kernel<1><<<dim3(8, 16), 256, 0, stream>>>(
      ab, wo16, (float*)d_out, 1024, 1024, nullptr, nullptr, nullptr);
}
